// GraphTransformer_21947282882994
// MI455X (gfx1250) — matrix operand order repaired and checked by reading, not compiled, not run on hardware
//
#include <hip/hip_runtime.h>
#include <hip/hip_bf16.h>
#include <math.h>
#include <stdint.h>

// Problem constants (match reference)
#define LAYERS 4
#define NN 10000
#define EE 40000
#define DD 256
#define HH 6
#define CC 256
#define HCx 1536
#define FFx 1024

typedef __attribute__((ext_vector_type(16))) __bf16 v16bf;
typedef __attribute__((ext_vector_type(8)))  float v8f;

union BF16x16 { uint32_t u[8]; v16bf v; };

// hardware packed f32->bf16 conversion (RNE): D[15:0]=bf16(lo), D[31:16]=bf16(hi)
__device__ __forceinline__ uint32_t pack2bf(float lo, float hi) {
  uint32_t r;
  asm("v_cvt_pk_bf16_f32 %0, %1, %2" : "=v"(r) : "v"(lo), "v"(hi));
  return r;
}

__device__ __forceinline__ float wave_sum(float s) {
  #pragma unroll
  for (int o = 16; o > 0; o >>= 1) s += __shfl_xor(s, o, 32);
  return s;
}

// monotone float<->uint encoding for atomicMax-based segment max
__device__ __forceinline__ unsigned fenc(float f) {
  unsigned b = __float_as_uint(f);
  return (b & 0x80000000u) ? ~b : (b | 0x80000000u);
}
__device__ __forceinline__ float fdec(unsigned k) {
  unsigned b = (k & 0x80000000u) ? (k & 0x7FFFFFFFu) : ~k;
  return __uint_as_float(b);
}
#define ENC_NEG_INF 0x007FFFFFu   // fenc(-inf)

// ---------------------------------------------------------------------------
// Tiled bf16 WMMA GEMM:  C[M,Nn] = act( A[M,K] @ B[K,Nn] + bias )
// block = 256 threads (8 waves), block tile 64x128, wave tile 32x32, K-step 32
// fp32 operands are converted to bf16 (v_cvt_pk_bf16_f32) while staging into
// LDS (coalesced float4 global loads; packed b128/b32 LDS stores; row-clamped
// bounds so EXEC stays full and no predicated load chains are generated).
// ---------------------------------------------------------------------------
__global__ __launch_bounds__(256)
void gemm_kernel(const float* __restrict__ A, const float* __restrict__ B,
                 const float* __restrict__ bias, float* __restrict__ Cd,
                 int M, int K, int Nn, int act) {
  __shared__ uint32_t As[64 * 16];   // [m][kpair]  (bf16 pair: lo=even k, hi=odd k)
  __shared__ uint32_t Bs[128 * 16];  // [n][kpair]
  const int tid  = threadIdx.x;
  const int lane = tid & 31;
  const int wave = tid >> 5;
  const int wr   = wave >> 2;        // 0..1 wave row
  const int wc   = wave & 3;         // 0..3 wave col
  const int m0   = blockIdx.y * 64;
  const int n0   = blockIdx.x * 128;
  const int half = lane >> 4;
  const int ml   = lane & 15;

  // staging thread mapping (coalesced along the fast dimension)
  const int ar  = tid >> 2;          // A row 0..63
  const int akb = (tid & 3) * 8;     // A k-base (8 floats per thread)
  const int bkp = tid >> 4;          // B k-pair 0..15
  const int bnb = (tid & 15) * 8;    // B n-base (8 floats per row, 2 rows)

  int gr = m0 + ar; if (gr >= M) gr = M - 1;   // clamp: rows>=M never stored

  v8f acc00 = {}, acc01 = {}, acc10 = {}, acc11 = {};

  for (int kk = 0; kk < K; kk += 32) {
    // ---- stage A (64x32): two float4 loads -> one ds_store_b128
    {
      const float4* ap = (const float4*)(A + (size_t)gr * K + kk + akb);
      float4 a0 = ap[0], a1 = ap[1];
      uint4 w;
      w.x = pack2bf(a0.x, a0.y); w.y = pack2bf(a0.z, a0.w);
      w.z = pack2bf(a1.x, a1.y); w.w = pack2bf(a1.z, a1.w);
      *(uint4*)&As[ar * 16 + (akb >> 1)] = w;
      if (kk + 32 < K) __builtin_prefetch(A + (size_t)gr * K + kk + 32 + akb, 0, 0);
    }
    // ---- stage B (32x128) -> transposed [n][kpair]
    {
      const int c0 = kk + 2 * bkp;
      const float4* bp0 = (const float4*)(B + (size_t)c0 * Nn + n0 + bnb);
      const float4* bp1 = (const float4*)(B + (size_t)(c0 + 1) * Nn + n0 + bnb);
      float4 r00 = bp0[0], r01 = bp0[1];
      float4 r10 = bp1[0], r11 = bp1[1];
      uint32_t p0 = pack2bf(r00.x, r10.x), p1 = pack2bf(r00.y, r10.y);
      uint32_t p2 = pack2bf(r00.z, r10.z), p3 = pack2bf(r00.w, r10.w);
      uint32_t p4 = pack2bf(r01.x, r11.x), p5 = pack2bf(r01.y, r11.y);
      uint32_t p6 = pack2bf(r01.z, r11.z), p7 = pack2bf(r01.w, r11.w);
      Bs[(bnb + 0) * 16 + bkp] = p0; Bs[(bnb + 1) * 16 + bkp] = p1;
      Bs[(bnb + 2) * 16 + bkp] = p2; Bs[(bnb + 3) * 16 + bkp] = p3;
      Bs[(bnb + 4) * 16 + bkp] = p4; Bs[(bnb + 5) * 16 + bkp] = p5;
      Bs[(bnb + 6) * 16 + bkp] = p6; Bs[(bnb + 7) * 16 + bkp] = p7;
      if (kk + 32 < K) __builtin_prefetch(B + (size_t)(c0 + 32) * Nn + n0 + bnb, 0, 0);
    }
    __syncthreads();

    // ---- build fragments per documented CDNA5 16-bit layouts (b128 LDS reads)
    BF16x16 a0f, a1f, b0f, b1f;
    const int arow0 = wr * 32 + ml;
    #pragma unroll
    for (int vv = 0; vv < 8; ++vv) {
      // u32 k-index: v<4 -> {0|4}+v ; v>=4 -> 8+{0|4}+(v-4)   ({} = lane half)
      int ku = ((vv < 4) ? 0 : 8) + half * 4 + (vv & 3);
      a0f.u[vv] = As[arow0 * 16 + ku];
      a1f.u[vv] = As[(arow0 + 16) * 16 + ku];
    }
    const int bcol0 = wc * 32 + ml;
    #pragma unroll
    for (int vv = 0; vv < 8; ++vv) {
      int ku = ((vv < 4) ? 0 : 8) + half * 4 + (vv & 3);   // same k order as A
      b0f.u[vv] = Bs[bcol0 * 16 + ku];
      b1f.u[vv] = Bs[(bcol0 + 16) * 16 + ku];
    }
    acc00 = __builtin_amdgcn_wmma_f32_16x16x32_bf16(false, a0f.v, false, b0f.v,
                                                    (short)0, acc00, false, false);
    acc01 = __builtin_amdgcn_wmma_f32_16x16x32_bf16(false, a0f.v, false, b1f.v,
                                                    (short)0, acc01, false, false);
    acc10 = __builtin_amdgcn_wmma_f32_16x16x32_bf16(false, a1f.v, false, b0f.v,
                                                    (short)0, acc10, false, false);
    acc11 = __builtin_amdgcn_wmma_f32_16x16x32_bf16(false, a1f.v, false, b1f.v,
                                                    (short)0, acc11, false, false);
    __syncthreads();
  }

  // ---- epilogue: C/D layout -> m_local = r + half*8
  #pragma unroll
  for (int rb = 0; rb < 2; ++rb) {
    #pragma unroll
    for (int cb = 0; cb < 2; ++cb) {
      v8f acc = (rb == 0) ? (cb == 0 ? acc00 : acc01)
                          : (cb == 0 ? acc10 : acc11);
      int nc = n0 + wc * 32 + cb * 16 + ml;
      float bv = bias ? bias[nc] : 0.0f;
      #pragma unroll
      for (int r = 0; r < 8; ++r) {
        int m = m0 + wr * 32 + rb * 16 + half * 8 + r;
        if (m >= M) continue;
        float v = acc[r] + bv;
        if (act == 1) v = 0.5f * v * (1.0f + erff(v * 0.70710678f));  // exact GELU
        Cd[(size_t)m * Nn + nc] = v;
      }
    }
  }
}

// ---------------------------------------------------------------------------
// LayerNorm over last dim (D=256). One wave per row.
// ---------------------------------------------------------------------------
__global__ __launch_bounds__(32)
void ln_kernel(const float* __restrict__ x, const float* __restrict__ g,
               const float* __restrict__ b, float* __restrict__ y, int rows) {
  int row = blockIdx.x;
  if (row >= rows) return;
  int lane = threadIdx.x;
  const float* xr = x + (size_t)row * DD;
  float vals[8];
  float s = 0.f;
  #pragma unroll
  for (int i = 0; i < 8; ++i) { vals[i] = xr[lane + i * 32]; s += vals[i]; }
  s = wave_sum(s);
  float mean = s * (1.0f / DD);
  float vs = 0.f;
  #pragma unroll
  for (int i = 0; i < 8; ++i) { float d = vals[i] - mean; vs += d * d; }
  vs = wave_sum(vs);
  float inv = rsqrtf(vs * (1.0f / DD) + 1e-5f);
  float* yr = y + (size_t)row * DD;
  #pragma unroll
  for (int i = 0; i < 8; ++i) {
    int c = lane + i * 32;
    yr[c] = (vals[i] - mean) * inv * g[c] + b[c];
  }
}

// ---------------------------------------------------------------------------
// Attention scratch init
// ---------------------------------------------------------------------------
__global__ void attn_init_kernel(unsigned* mkey, float* denom, float* attout) {
  int idx = blockIdx.x * blockDim.x + threadIdx.x;
  if (idx < NN * HH) { mkey[idx] = ENC_NEG_INF; denom[idx] = 0.f; }
  if (idx < NN * DD) attout[idx] = 0.f;
}

// one wave per (edge, head): alpha = <q[dst], k[src]+e> * scale
__global__ __launch_bounds__(256)
void alpha_kernel(const float* __restrict__ q, const float* __restrict__ k,
                  const float* __restrict__ ep, const int* __restrict__ ei,
                  float* __restrict__ alpha) {
  int wave = threadIdx.x >> 5, lane = threadIdx.x & 31;
  int eh = blockIdx.x * 8 + wave;
  if (eh >= EE * HH) return;
  int e = eh / HH, h = eh % HH;
  int s = ei[e], d = ei[EE + e];
  const float* qr = q + (size_t)d * HCx + h * CC;
  const float* kr = k + (size_t)s * HCx + h * CC;
  const float* er = ep + (size_t)e * HCx + h * CC;
  float acc = 0.f;
  #pragma unroll
  for (int i = 0; i < 8; ++i) {
    int c = lane + i * 32;
    acc += qr[c] * (kr[c] + er[c]);
  }
  acc = wave_sum(acc);
  if (lane == 0) alpha[eh] = acc * 0.0625f;  // 1/sqrt(256)
}

__global__ void segmax_kernel(const float* __restrict__ alpha,
                              const int* __restrict__ ei, unsigned* __restrict__ mkey) {
  int eh = blockIdx.x * blockDim.x + threadIdx.x;
  if (eh >= EE * HH) return;
  int e = eh / HH, h = eh % HH;
  int d = ei[EE + e];
  atomicMax(&mkey[d * HH + h], fenc(alpha[eh]));
}

__global__ void expsum_kernel(const float* __restrict__ alpha,
                              const unsigned* __restrict__ mkey,
                              const int* __restrict__ ei,
                              float* __restrict__ aexp, float* __restrict__ denom) {
  int eh = blockIdx.x * blockDim.x + threadIdx.x;
  if (eh >= EE * HH) return;
  int e = eh / HH, h = eh % HH;
  int d = ei[EE + e];
  float a = expf(alpha[eh] - fdec(mkey[d * HH + h]));
  aexp[eh] = a;
  atomicAdd(&denom[d * HH + h], a);
}

// one wave per (edge, head): attout[dst] += (v[src]+e) * a/denom / H
__global__ __launch_bounds__(256)
void scatter_kernel(const float* __restrict__ v, const float* __restrict__ ep,
                    const float* __restrict__ aexp, const float* __restrict__ denom,
                    const int* __restrict__ ei, float* __restrict__ attout) {
  int wave = threadIdx.x >> 5, lane = threadIdx.x & 31;
  int eh = blockIdx.x * 8 + wave;
  if (eh >= EE * HH) return;
  int e = eh / HH, h = eh % HH;
  int s = ei[e], d = ei[EE + e];
  float coeff = aexp[eh] / (denom[d * HH + h] + 1e-16f) * (1.0f / HH);
  const float* vr = v + (size_t)s * HCx + h * CC;
  const float* er = ep + (size_t)e * HCx + h * CC;
  float* outr = attout + (size_t)d * DD;
  #pragma unroll
  for (int i = 0; i < 8; ++i) {
    int c = lane + i * 32;
    atomicAdd(&outr[c], (vr[c] + er[c]) * coeff);
  }
}

// beta-gated residual: nodes = beta*x_r + (1-beta)*out
__global__ __launch_bounds__(256)
void beta_kernel(const float* __restrict__ attout, const float* __restrict__ xr,
                 const float* __restrict__ Wb, float* __restrict__ nodes) {
  int wave = threadIdx.x >> 5, lane = threadIdx.x & 31;
  int n = blockIdx.x * 8 + wave;
  if (n >= NN) return;
  const float* o = attout + (size_t)n * DD;
  const float* r = xr + (size_t)n * CC;
  float ov[8], rv[8], s = 0.f;
  #pragma unroll
  for (int i = 0; i < 8; ++i) {
    int c = lane + i * 32;
    ov[i] = o[c]; rv[i] = r[c];
    s += ov[i] * Wb[c] + rv[i] * Wb[CC + c] + (ov[i] - rv[i]) * Wb[2 * CC + c];
  }
  s = wave_sum(s);
  float beta = 1.0f / (1.0f + expf(-s));
  float* out = nodes + (size_t)n * DD;
  #pragma unroll
  for (int i = 0; i < 8; ++i) {
    int c = lane + i * 32;
    out[c] = beta * rv[i] + (1.0f - beta) * ov[i];
  }
}

// FFN gated residual: x = h*g + x*(1-g), g = sigmoid([h,x,h-x] @ Wg)
__global__ __launch_bounds__(256)
void gate_kernel(float* __restrict__ x, const float* __restrict__ h,
                 const float* __restrict__ Wg, int rows) {
  int wave = threadIdx.x >> 5, lane = threadIdx.x & 31;
  int n = blockIdx.x * 8 + wave;
  if (n >= rows) return;
  float* xr = x + (size_t)n * DD;
  const float* hr = h + (size_t)n * DD;
  float xv[8], hv[8], s = 0.f;
  #pragma unroll
  for (int i = 0; i < 8; ++i) {
    int c = lane + i * 32;
    xv[i] = xr[c]; hv[i] = hr[c];
    s += hv[i] * Wg[c] + xv[i] * Wg[DD + c] + (hv[i] - xv[i]) * Wg[2 * DD + c];
  }
  s = wave_sum(s);
  float g = 1.0f / (1.0f + expf(-s));
  #pragma unroll
  for (int i = 0; i < 8; ++i) {
    int c = lane + i * 32;
    xr[c] = hv[i] * g + xv[i] * (1.0f - g);
  }
}

// ---------------------------------------------------------------------------
extern "C" void kernel_launch(void* const* d_in, const int* in_sizes, int n_in,
                              void* d_out, int out_size, void* d_ws, size_t ws_size,
                              hipStream_t stream) {
  // inputs in setup_inputs() order
  const float* in_nodes  = (const float*)d_in[0];
  const float* in_edges  = (const float*)d_in[1];
  const int*   ei        = (const int*)d_in[2];
  const float* ln_attn_g = (const float*)d_in[3];
  const float* ln_attn_b = (const float*)d_in[4];
  const float* Wq = (const float*)d_in[5];  const float* bq = (const float*)d_in[6];
  const float* Wk = (const float*)d_in[7];  const float* bk = (const float*)d_in[8];
  const float* Wv = (const float*)d_in[9];  const float* bv = (const float*)d_in[10];
  const float* We = (const float*)d_in[11];
  const float* Wskip = (const float*)d_in[12]; const float* bskip = (const float*)d_in[13];
  const float* Wbeta = (const float*)d_in[14];
  const float* ln_nf_g = (const float*)d_in[15]; const float* ln_nf_b = (const float*)d_in[16];
  const float* Wn1 = (const float*)d_in[17]; const float* bn1 = (const float*)d_in[18];
  const float* Wn2 = (const float*)d_in[19]; const float* bn2 = (const float*)d_in[20];
  const float* Wng = (const float*)d_in[21];
  const float* ln_ef_g = (const float*)d_in[22]; const float* ln_ef_b = (const float*)d_in[23];
  const float* We1 = (const float*)d_in[24]; const float* be1 = (const float*)d_in[25];
  const float* We2 = (const float*)d_in[26]; const float* be2 = (const float*)d_in[27];
  const float* Weg = (const float*)d_in[28];

  // workspace layout (fp32 elements)
  float* ws = (float*)d_ws;
  size_t off = 0;
  auto alloc = [&](size_t n) { float* p = ws + off; off += n; return p; };
  float* nodes_buf = alloc((size_t)NN * DD);
  float* edges_buf = alloc((size_t)EE * DD);
  float* xn        = alloc((size_t)NN * DD);
  float* qb        = alloc((size_t)NN * HCx);
  float* kb2       = alloc((size_t)NN * HCx);
  float* vb        = alloc((size_t)NN * HCx);
  float* eproj     = alloc((size_t)EE * HCx);
  float* alpha     = alloc((size_t)EE * HH);
  float* aexp      = alloc((size_t)EE * HH);
  float* denom     = alloc((size_t)NN * HH);
  unsigned* mkey   = (unsigned*)alloc((size_t)NN * HH);
  float* attout    = alloc((size_t)NN * DD);
  float* xr        = alloc((size_t)NN * CC);
  float* en        = alloc((size_t)EE * DD);
  float* h1        = alloc((size_t)EE * FFx);   // shared node/edge hidden
  float* h2        = alloc((size_t)EE * DD);

  hipMemcpyAsync(nodes_buf, in_nodes, (size_t)NN * DD * 4, hipMemcpyDeviceToDevice, stream);
  hipMemcpyAsync(edges_buf, in_edges, (size_t)EE * DD * 4, hipMemcpyDeviceToDevice, stream);

  auto gemm = [&](const float* A, const float* B, const float* bias, float* C,
                  int M, int K, int Nn, int act) {
    dim3 grid((Nn + 127) / 128, (M + 63) / 64);
    gemm_kernel<<<grid, 256, 0, stream>>>(A, B, bias, C, M, K, Nn, act);
  };

  const int ehBlocks8   = (EE * HH + 7) / 8;
  const int ehBlocks256 = (EE * HH + 255) / 256;

  for (int l = 0; l < LAYERS; ++l) {
    const size_t wQ = (size_t)l * DD * HCx, wF1 = (size_t)l * DD * FFx, wF2 = (size_t)l * FFx * DD;
    // --- attention block ---
    ln_kernel<<<NN, 32, 0, stream>>>(nodes_buf, ln_attn_g + l * DD, ln_attn_b + l * DD, xn, NN);
    gemm(xn, Wq + wQ, bq + (size_t)l * HCx, qb, NN, DD, HCx, 0);
    gemm(xn, Wk + wQ, bk + (size_t)l * HCx, kb2, NN, DD, HCx, 0);
    gemm(xn, Wv + wQ, bv + (size_t)l * HCx, vb, NN, DD, HCx, 0);
    gemm(edges_buf, We + wQ, nullptr, eproj, EE, DD, HCx, 0);
    attn_init_kernel<<<(NN * DD + 255) / 256, 256, 0, stream>>>(mkey, denom, attout);
    alpha_kernel<<<ehBlocks8, 256, 0, stream>>>(qb, kb2, eproj, ei, alpha);
    segmax_kernel<<<ehBlocks256, 256, 0, stream>>>(alpha, ei, mkey);
    expsum_kernel<<<ehBlocks256, 256, 0, stream>>>(alpha, mkey, ei, aexp, denom);
    scatter_kernel<<<ehBlocks8, 256, 0, stream>>>(vb, eproj, aexp, denom, ei, attout);
    gemm(xn, Wskip + (size_t)l * DD * CC, bskip + (size_t)l * CC, xr, NN, DD, CC, 0);
    beta_kernel<<<(NN + 7) / 8, 256, 0, stream>>>(attout, xr, Wbeta + (size_t)l * 3 * CC, nodes_buf);
    // --- node FFN (gated residual) ---
    ln_kernel<<<NN, 32, 0, stream>>>(nodes_buf, ln_nf_g + l * DD, ln_nf_b + l * DD, xn, NN);
    gemm(xn, Wn1 + wF1, bn1 + (size_t)l * FFx, h1, NN, DD, FFx, 1);
    gemm(h1, Wn2 + wF2, bn2 + (size_t)l * DD, h2, NN, FFx, DD, 0);
    gate_kernel<<<(NN + 7) / 8, 256, 0, stream>>>(nodes_buf, h2, Wng + (size_t)l * 3 * DD, NN);
    // --- edge FFN (gated residual) ---
    ln_kernel<<<EE, 32, 0, stream>>>(edges_buf, ln_ef_g + l * DD, ln_ef_b + l * DD, en, EE);
    gemm(en, We1 + wF1, be1 + (size_t)l * FFx, h1, EE, DD, FFx, 1);
    gemm(h1, We2 + wF2, be2 + (size_t)l * DD, h2, EE, FFx, DD, 0);
    gate_kernel<<<(EE + 7) / 8, 256, 0, stream>>>(edges_buf, h2, Weg + (size_t)l * 3 * DD, EE);
  }

  hipMemcpyAsync(d_out, nodes_buf, (size_t)NN * DD * 4, hipMemcpyDeviceToDevice, stream);
}
